// MultiHeadAttention_47459388621648
// MI455X (gfx1250) — matrix operand order repaired and checked by reading, not compiled, not run on hardware
//
#include <hip/hip_runtime.h>
#include <hip/hip_bf16.h>

// ---------------- problem constants (match reference) ----------------
constexpr int kD  = 512;   // d_encoder
constexpr int kH  = 8;     // heads
constexpr int kDH = 64;    // d_attn per head
constexpr int kB  = 8;
constexpr int kS  = 1024;

typedef __attribute__((ext_vector_type(16))) _Float16 v16h;
typedef __attribute__((ext_vector_type(8)))  float    v8f;

union v16h_u {
  v16h v;
  uint4 q[2];
};

// ---------------- WMMA helpers ----------------
__device__ inline v8f wmma16(v16h a, v16h b, v8f c) {
  // D = A(16x32 f16) * B(32x16 f16) + C(16x16 f32)
  return __builtin_amdgcn_wmma_f32_16x16x32_f16(
      /*neg_a=*/false, a, /*neg_b=*/false, b,
      /*c_mod=*/(short)0, c, /*reuse_a=*/false, /*reuse_b=*/false);
}

// A-fragment (16x32 f16): lane holds row M=lane&15; lanes 0-15 take K
// [kk+0..7] and [kk+16..23]; lanes 16-31 take [kk+8..15] and [kk+24..31].
__device__ inline v16h load_a_frag(const _Float16* __restrict__ row, int kk, int lane) {
  int base = kk + ((lane & 16) ? 8 : 0);
  v16h_u r;
  r.q[0] = *(const uint4*)(row + base);
  r.q[1] = *(const uint4*)(row + base + 16);
  return r.v;
}

// Same A-fragment but converting from an f32 source row.
__device__ inline v16h load_a_frag_f32(const float* __restrict__ row, int kk, int lane) {
  int base = kk + ((lane & 16) ? 8 : 0);
  float4 a0 = *(const float4*)(row + base);
  float4 a1 = *(const float4*)(row + base + 4);
  float4 b0 = *(const float4*)(row + base + 16);
  float4 b1 = *(const float4*)(row + base + 20);
  v16h r;
  r[0] = (_Float16)a0.x; r[1]  = (_Float16)a0.y; r[2]  = (_Float16)a0.z; r[3]  = (_Float16)a0.w;
  r[4] = (_Float16)a1.x; r[5]  = (_Float16)a1.y; r[6]  = (_Float16)a1.z; r[7]  = (_Float16)a1.w;
  r[8] = (_Float16)b0.x; r[9]  = (_Float16)b0.y; r[10] = (_Float16)b0.z; r[11] = (_Float16)b0.w;
  r[12] = (_Float16)b1.x; r[13] = (_Float16)b1.y; r[14] = (_Float16)b1.z; r[15] = (_Float16)b1.w;
  return r;
}

// B-fragment (32x16 f16): lane holds column N=lane&15; lanes 0-15 take K
// [kk+0..7] and [kk+16..23]; lanes 16-31 take [kk+8..15] and [kk+24..31]. `col` points at
// contiguous K values for this lane's column (i.e. B is stored K-contiguous per column).
__device__ inline v16h load_b_frag(const _Float16* __restrict__ col, int kk, int lane) {
  int kb = kk + ((lane & 16) ? 8 : 0);
  v16h_u r;
  r.q[0] = *(const uint4*)(col + kb);
  r.q[1] = *(const uint4*)(col + kb + 16);
  return r.v;
}

// ---------------- stage 0: weight convert + transpose to f16 ----------------
// W: [NH, Dd, E] f32 -> WT: [NH, E, Dd] f16 (so B-fragments are contiguous)
__global__ void cvt_transpose_kernel(const float* __restrict__ W, _Float16* __restrict__ WT,
                                     int NH, int Dd, int E) {
  int idx = blockIdx.x * blockDim.x + threadIdx.x;
  int total = NH * Dd * E;
  if (idx >= total) return;
  int e = idx % E;
  int d = (idx / E) % Dd;
  int hh = idx / (E * Dd);
  WT[((size_t)hh * E + e) * Dd + d] = (_Float16)W[idx];
}

// ---------------- stage 1: per-head projections ----------------
// One wave computes a FULL 16x64 output tile (16 seq rows x all DH columns):
// one shared A-fragment feeds 4 WMMAs per k-step. All 5 fragments of a
// k-step are loaded into distinct locals BEFORE the WMMAs so the backend can
// batch the global_load_b128s in one clause and stagger the loadcnt waits.
__global__ __launch_bounds__(32) void proj_kernel(
    const float* __restrict__ Q, const float* __restrict__ K, const float* __restrict__ V,
    const _Float16* __restrict__ WqT, const _Float16* __restrict__ WkT,
    const _Float16* __restrict__ WvT,
    const float* __restrict__ bq, const float* __restrict__ bk, const float* __restrict__ bv,
    _Float16* __restrict__ Qh, _Float16* __restrict__ Kh, _Float16* __restrict__ Vt) {
  int lane = threadIdx.x;
  int lm = lane & 15, hi = (lane >> 4) & 1;
  int id = blockIdx.x;
  int stile = id & 63; id >>= 6;   // S/16
  int h     = id & 7;  id >>= 3;
  int b     = id & 7;  id >>= 3;
  int which = id;                  // 0=Q 1=K 2=V

  const float*    X    = (which == 0) ? Q   : (which == 1) ? K   : V;
  const _Float16* WT   = (which == 0) ? WqT : (which == 1) ? WkT : WvT;
  const float*    bias = (which == 0) ? bq  : (which == 1) ? bk  : bv;

  const float*    xrow = X + ((size_t)b * kS + stile * 16 + lm) * kD;
  const _Float16* wc0  = WT + ((size_t)h * kDH + 0 * 16 + lm) * kD;
  const _Float16* wc1  = WT + ((size_t)h * kDH + 1 * 16 + lm) * kD;
  const _Float16* wc2  = WT + ((size_t)h * kDH + 2 * 16 + lm) * kD;
  const _Float16* wc3  = WT + ((size_t)h * kDH + 3 * 16 + lm) * kD;

  v8f c0 = {}, c1 = {}, c2 = {}, c3 = {};
#pragma unroll 2
  for (int k = 0; k < kD; k += 32) {
    // batch all loads of this k-step first
    v16h a  = load_a_frag_f32(xrow, k, lane);
    v16h b0 = load_b_frag(wc0, k, lane);
    v16h b1 = load_b_frag(wc1, k, lane);
    v16h b2 = load_b_frag(wc2, k, lane);
    v16h b3 = load_b_frag(wc3, k, lane);
    c0 = wmma16(a, b0, c0);
    c1 = wmma16(a, b1, c1);
    c2 = wmma16(a, b2, c2);
    c3 = wmma16(a, b3, c3);
  }

  float bb0 = bias[h * kDH + 0 * 16 + lm];
  float bb1 = bias[h * kDH + 1 * 16 + lm];
  float bb2 = bias[h * kDH + 2 * 16 + lm];
  float bb3 = bias[h * kDH + 3 * 16 + lm];
  size_t bh = (size_t)b * kH + h;
  if (which == 2) {
    _Float16* dst = Vt + bh * (size_t)kDH * kS;  // [DH][S]
#pragma unroll
    for (int r = 0; r < 8; r++) {
      int row = stile * 16 + r + 8 * hi;
      dst[(size_t)(0 * 16 + lm) * kS + row] = (_Float16)(c0[r] + bb0);
      dst[(size_t)(1 * 16 + lm) * kS + row] = (_Float16)(c1[r] + bb1);
      dst[(size_t)(2 * 16 + lm) * kS + row] = (_Float16)(c2[r] + bb2);
      dst[(size_t)(3 * 16 + lm) * kS + row] = (_Float16)(c3[r] + bb3);
    }
  } else {
    _Float16* dst = ((which == 0) ? Qh : Kh) + bh * (size_t)kS * kDH;  // [S][DH]
#pragma unroll
    for (int r = 0; r < 8; r++) {
      int row = stile * 16 + r + 8 * hi;
      _Float16* orow = dst + (size_t)row * kDH;
      orow[0 * 16 + lm] = (_Float16)(c0[r] + bb0);
      orow[1 * 16 + lm] = (_Float16)(c1[r] + bb1);
      orow[2 * 16 + lm] = (_Float16)(c2[r] + bb2);
      orow[3 * 16 + lm] = (_Float16)(c3[r] + bb3);
    }
  }
}

// ---------------- stage 2: fused flash attention ----------------
// One wave owns 16 query rows of one (b,h). Streams keys 32 at a time:
// scores via 4 WMMA, online softmax (f32), P staged through LDS to convert
// C-fragment layout -> A-fragment layout, then 4 WMMA against Vt tiles.
// Never materializes the 256MB score tensor.
__global__ __launch_bounds__(32) void attn_kernel(
    const _Float16* __restrict__ Qh, const _Float16* __restrict__ Kh,
    const _Float16* __restrict__ Vt, _Float16* __restrict__ Xh) {
  __shared__ __align__(16) _Float16 Plds[16 * 32];
  int lane = threadIdx.x;
  int lm = lane & 15, hi = (lane >> 4) & 1;
  int qi = blockIdx.x & 63;             // S/16
  size_t bh = blockIdx.x >> 6;          // b*H + h
  const _Float16* Qbh = Qh + bh * (size_t)kS * kDH;
  const _Float16* Kbh = Kh + bh * (size_t)kS * kDH;
  const _Float16* Vbh = Vt + bh * (size_t)kDH * kS;
  _Float16*       Xbh = Xh + bh * (size_t)kS * kDH;

  const _Float16* qrow = Qbh + (size_t)(qi * 16 + lm) * kDH;
  v16h aQ0 = load_a_frag(qrow, 0, lane);
  v16h aQ1 = load_a_frag(qrow, 32, lane);

  float m[8], l[8], alpha[8];
  v8f O0 = {}, O1 = {}, O2 = {}, O3 = {};
#pragma unroll
  for (int r = 0; r < 8; r++) { m[r] = -1e30f; l[r] = 0.0f; }
  const float scale = 0.125f;  // 1/sqrt(64)

  for (int j = 0; j < kS / 16; j += 2) {
    // -- scores: two 16x16 tiles over 32 keys; batch the 4 K-fragment loads --
    const _Float16* k0 = Kbh + (size_t)(j * 16 + lm) * kDH;
    const _Float16* k1 = k0 + 16 * kDH;
    v16h bk00 = load_b_frag(k0, 0, lane);
    v16h bk01 = load_b_frag(k0, 32, lane);
    v16h bk10 = load_b_frag(k1, 0, lane);
    v16h bk11 = load_b_frag(k1, 32, lane);
    v8f s0 = {}, s1 = {};
    s0 = wmma16(aQ0, bk00, s0);
    s0 = wmma16(aQ1, bk01, s0);
    s1 = wmma16(aQ0, bk10, s1);
    s1 = wmma16(aQ1, bk11, s1);
    if (j + 2 < kS / 16) {
      __builtin_prefetch(Kbh + (size_t)((j + 2) * 16 + lm) * kDH, 0, 1);
      __builtin_prefetch(Vbh + (size_t)lm * kS + (j + 2) * 16, 0, 1);
    }

    // -- online softmax; row values live across 16-lane halves --
#pragma unroll
    for (int r = 0; r < 8; r++) {
      float v0 = s0[r] * scale;
      float v1 = s1[r] * scale;
      float mx = fmaxf(v0, v1);
      mx = fmaxf(mx, __shfl_xor(mx, 1, 32));
      mx = fmaxf(mx, __shfl_xor(mx, 2, 32));
      mx = fmaxf(mx, __shfl_xor(mx, 4, 32));
      mx = fmaxf(mx, __shfl_xor(mx, 8, 32));
      float mn = fmaxf(m[r], mx);
      float p0 = __expf(v0 - mn);
      float p1 = __expf(v1 - mn);
      float rs = p0 + p1;
      rs += __shfl_xor(rs, 1, 32);
      rs += __shfl_xor(rs, 2, 32);
      rs += __shfl_xor(rs, 4, 32);
      rs += __shfl_xor(rs, 8, 32);
      alpha[r] = __expf(m[r] - mn);
      l[r] = l[r] * alpha[r] + rs;
      m[r] = mn;
      // P tile (16x32 f16) to LDS, row-major, stride 32
      Plds[(r + 8 * hi) * 32 + lm]      = (_Float16)p0;
      Plds[(r + 8 * hi) * 32 + 16 + lm] = (_Float16)p1;
    }
#pragma unroll
    for (int r = 0; r < 8; r++) {
      O0[r] *= alpha[r]; O1[r] *= alpha[r]; O2[r] *= alpha[r]; O3[r] *= alpha[r];
    }
    __syncthreads();

    // -- O += P(16x32) x V(32x64); batch V-fragment loads + LDS P read --
    const _Float16* vb = Vbh + (size_t)lm * kS + j * 16;
    v16h bv0 = load_b_frag(vb + 0 * 16 * kS, 0, lane);
    v16h bv1 = load_b_frag(vb + 1 * 16 * kS, 0, lane);
    v16h bv2 = load_b_frag(vb + 2 * 16 * kS, 0, lane);
    v16h bv3 = load_b_frag(vb + 3 * 16 * kS, 0, lane);
    v16h aP  = load_a_frag(&Plds[lm * 32], 0, lane);
    O0 = wmma16(aP, bv0, O0);
    O1 = wmma16(aP, bv1, O1);
    O2 = wmma16(aP, bv2, O2);
    O3 = wmma16(aP, bv3, O3);
    __syncthreads();
  }

  // -- normalize and store x (f16, [S][DH] per (b,h)) --
#pragma unroll
  for (int r = 0; r < 8; r++) {
    float inv = 1.0f / l[r];
    int row = qi * 16 + r + 8 * hi;
    _Float16* orow = Xbh + (size_t)row * kDH;
    orow[0 * 16 + lm] = (_Float16)(O0[r] * inv);
    orow[1 * 16 + lm] = (_Float16)(O1[r] * inv);
    orow[2 * 16 + lm] = (_Float16)(O2[r] * inv);
    orow[3 * 16 + lm] = (_Float16)(O3[r] * inv);
  }
}

// ---------------- stage 3: output projection ----------------
// Reads the [B,H,S,DH] buffer FLAT as [B*S, 512], faithfully replicating the
// reference's layout-reinterpreting reshape. out = X @ Wo + bo (f32).
// One wave computes a 16x64 tile; fragments batched before the 4 WMMAs.
__global__ __launch_bounds__(32) void out_proj_kernel(
    const _Float16* __restrict__ X, const _Float16* __restrict__ WoT,
    const float* __restrict__ bo, float* __restrict__ out) {
  int lane = threadIdx.x;
  int lm = lane & 15, hi = (lane >> 4) & 1;
  int ng    = blockIdx.x & 7;    // D/64 column groups
  int mtile = blockIdx.x >> 3;   // (B*S)/16
  const _Float16* arow = X + (size_t)(mtile * 16 + lm) * kD;
  const _Float16* bc0  = WoT + (size_t)(ng * 64 + 0 * 16 + lm) * kD;
  const _Float16* bc1  = WoT + (size_t)(ng * 64 + 1 * 16 + lm) * kD;
  const _Float16* bc2  = WoT + (size_t)(ng * 64 + 2 * 16 + lm) * kD;
  const _Float16* bc3  = WoT + (size_t)(ng * 64 + 3 * 16 + lm) * kD;
  v8f c0 = {}, c1 = {}, c2 = {}, c3 = {};
#pragma unroll 2
  for (int k = 0; k < kD; k += 32) {
    v16h a  = load_a_frag(arow, k, lane);
    v16h b0 = load_b_frag(bc0, k, lane);
    v16h b1 = load_b_frag(bc1, k, lane);
    v16h b2 = load_b_frag(bc2, k, lane);
    v16h b3 = load_b_frag(bc3, k, lane);
    c0 = wmma16(a, b0, c0);
    c1 = wmma16(a, b1, c1);
    c2 = wmma16(a, b2, c2);
    c3 = wmma16(a, b3, c3);
  }
  float bb0 = bo[ng * 64 + 0 * 16 + lm];
  float bb1 = bo[ng * 64 + 1 * 16 + lm];
  float bb2 = bo[ng * 64 + 2 * 16 + lm];
  float bb3 = bo[ng * 64 + 3 * 16 + lm];
#pragma unroll
  for (int r = 0; r < 8; r++) {
    int row = mtile * 16 + r + 8 * hi;
    float* orow = out + (size_t)row * kD + ng * 64;
    orow[0 * 16 + lm] = c0[r] + bb0;
    orow[1 * 16 + lm] = c1[r] + bb1;
    orow[2 * 16 + lm] = c2[r] + bb2;
    orow[3 * 16 + lm] = c3[r] + bb3;
  }
}

// ---------------- launch ----------------
extern "C" void kernel_launch(void* const* d_in, const int* in_sizes, int n_in,
                              void* d_out, int out_size, void* d_ws, size_t ws_size,
                              hipStream_t stream) {
  const float* Q  = (const float*)d_in[0];
  const float* K  = (const float*)d_in[1];
  const float* V  = (const float*)d_in[2];
  const float* Wq = (const float*)d_in[3];
  const float* bq = (const float*)d_in[4];
  const float* Wk = (const float*)d_in[5];
  const float* bk = (const float*)d_in[6];
  const float* Wv = (const float*)d_in[7];
  const float* bv = (const float*)d_in[8];
  const float* Wo = (const float*)d_in[9];
  const float* bo = (const float*)d_in[10];
  (void)in_sizes; (void)n_in; (void)out_size; (void)ws_size;

  // workspace carve-up (bytes)
  char* ws = (char*)d_ws;
  const size_t szBHSE = (size_t)kB * kH * kS * kDH * sizeof(_Float16);  // 8 MB
  const size_t szWT   = (size_t)kH * kD * kDH * sizeof(_Float16);       // 512 KB
  _Float16* Qh16 = (_Float16*)(ws + 0 * szBHSE);
  _Float16* Kh16 = (_Float16*)(ws + 1 * szBHSE);
  _Float16* Vt16 = (_Float16*)(ws + 2 * szBHSE);
  _Float16* Xh16 = (_Float16*)(ws + 3 * szBHSE);
  _Float16* WqT  = (_Float16*)(ws + 4 * szBHSE);
  _Float16* WkT  = (_Float16*)(ws + 4 * szBHSE + 1 * szWT);
  _Float16* WvT  = (_Float16*)(ws + 4 * szBHSE + 2 * szWT);
  _Float16* WoT  = (_Float16*)(ws + 4 * szBHSE + 3 * szWT);

  // stage 0: weight conversions
  {
    int n = kH * kD * kDH;
    int g = (n + 255) / 256;
    cvt_transpose_kernel<<<g, 256, 0, stream>>>(Wq, WqT, kH, kD, kDH);
    cvt_transpose_kernel<<<g, 256, 0, stream>>>(Wk, WkT, kH, kD, kDH);
    cvt_transpose_kernel<<<g, 256, 0, stream>>>(Wv, WvT, kH, kD, kDH);
    int n2 = kD * kD;
    cvt_transpose_kernel<<<(n2 + 255) / 256, 256, 0, stream>>>(Wo, WoT, 1, kD, kD);
  }

  // stage 1: Q/K/V per-head projections (3 * B * H * S/16 waves, 16x64 each)
  {
    int grid = 3 * kB * kH * (kS / 16);
    proj_kernel<<<grid, 32, 0, stream>>>(Q, K, V, WqT, WkT, WvT, bq, bk, bv,
                                         Qh16, Kh16, Vt16);
  }

  // stage 2: fused attention (B * H * S/16 waves)
  attn_kernel<<<kB * kH * (kS / 16), 32, 0, stream>>>(Qh16, Kh16, Vt16, Xh16);

  // stage 3: output projection ((B*S/16) * (D/64) waves, 16x64 each)
  out_proj_kernel<<<(kB * kS / 16) * (kD / 64), 32, 0, stream>>>(
      Xh16, WoT, bo, (float*)d_out);
}
